// YatNMN_48610439856164
// MI455X (gfx1250) — compile-verified, matrix operand order checked by reading, not run on hardware
//
#include <hip/hip_runtime.h>
#include <math.h>

#define M_TOT 8192
#define N_TOT 8192
#define K_TOT 2048
#define EPSV  1e-5f

#define BM 128
#define BN 128
#define BK 32
#define NTILES (K_TOT / BK)

typedef __bf16 bf16_t;
typedef __bf16 v16bf __attribute__((ext_vector_type(16)));
typedef __bf16 v8bf  __attribute__((ext_vector_type(8)));
typedef __bf16 v4bf  __attribute__((ext_vector_type(4)));
typedef float  v8f   __attribute__((ext_vector_type(8)));
typedef float  v4f   __attribute__((ext_vector_type(4)));

// ---- CDNA5 async copy: 16B global -> LDS, tracked by ASYNCcnt ----
__device__ __forceinline__ void async_copy16(const bf16_t* g, const bf16_t* l) {
  unsigned lds = (unsigned)(size_t)l;  // LDS aperture: low 32 bits = LDS byte address
  asm volatile("global_load_async_to_lds_b128 %0, %1, off"
               :: "v"(lds), "v"(g) : "memory");
}

// =================================================================
// Pre-pass 1: row squared-norm (exact f32) fused with X -> hi/lo bf16
// =================================================================
__global__ __launch_bounds__(256) void yat_rownorm_convert(const float* __restrict__ X,
                                                           bf16_t* __restrict__ Xh,
                                                           bf16_t* __restrict__ Xl,
                                                           float* __restrict__ x2) {
  const int row = blockIdx.x;
  const float* p = X + (size_t)row * K_TOT;
  bf16_t* ph = Xh + (size_t)row * K_TOT;
  bf16_t* pl = Xl + (size_t)row * K_TOT;
  float s = 0.0f;
  for (int i = threadIdx.x; i < K_TOT / 4; i += 256) {
    v4f v = ((const v4f*)p)[i];
    v4bf h, l;
    #pragma unroll
    for (int j = 0; j < 4; ++j) {
      bf16_t hh = (bf16_t)v[j];
      h[j] = hh;
      l[j] = (bf16_t)(v[j] - (float)hh);
      s += v[j] * v[j];
    }
    ((v4bf*)ph)[i] = h;
    ((v4bf*)pl)[i] = l;
  }
  for (int off = 16; off > 0; off >>= 1) s += __shfl_down(s, off, 32);
  __shared__ float ws[8];
  if ((threadIdx.x & 31) == 0) ws[threadIdx.x >> 5] = s;
  __syncthreads();
  if (threadIdx.x == 0) {
    float t = 0.0f;
    #pragma unroll
    for (int i = 0; i < 8; ++i) t += ws[i];
    x2[row] = t;
  }
}

// plain row-norm (fallback path, no conversion)
__global__ __launch_bounds__(256) void yat_row_sq(const float* __restrict__ X,
                                                  float* __restrict__ x2) {
  const int row = blockIdx.x;
  const float* p = X + (size_t)row * K_TOT;
  float s = 0.0f;
  for (int i = threadIdx.x; i < K_TOT / 4; i += 256) {
    v4f v = ((const v4f*)p)[i];
    s += v[0]*v[0] + v[1]*v[1] + v[2]*v[2] + v[3]*v[3];
  }
  for (int off = 16; off > 0; off >>= 1) s += __shfl_down(s, off, 32);
  __shared__ float ws[8];
  if ((threadIdx.x & 31) == 0) ws[threadIdx.x >> 5] = s;
  __syncthreads();
  if (threadIdx.x == 0) {
    float t = 0.0f;
    #pragma unroll
    for (int i = 0; i < 8; ++i) t += ws[i];
    x2[row] = t;
  }
}

// =================================================================
// Pre-pass 2: col squared-norm (exact f32, from original W)
// =================================================================
__global__ __launch_bounds__(256) void yat_col_sq(const float* __restrict__ W,
                                                  float* __restrict__ w2) {
  const int col = blockIdx.x * 256 + threadIdx.x;
  float s = 0.0f;
  for (int k = 0; k < K_TOT; ++k) {
    float v = W[(size_t)k * N_TOT + col];
    s += v * v;
  }
  w2[col] = s;
}

// =================================================================
// Pre-pass 3: W[K][N] -> Wt[N][K] transposed hi/lo bf16 (tiled via LDS)
// =================================================================
__global__ __launch_bounds__(256) void yat_transpose_convert(const float* __restrict__ W,
                                                             bf16_t* __restrict__ Wth,
                                                             bf16_t* __restrict__ Wtl) {
  __shared__ float T[32][33];
  const int n0 = blockIdx.x * 32;
  const int k0 = blockIdx.y * 32;
  const int tn = threadIdx.x & 31;
  const int tk = threadIdx.x >> 5;  // 0..7
  #pragma unroll
  for (int p = 0; p < 4; ++p) {
    const int k = tk + p * 8;
    T[k][tn] = W[(size_t)(k0 + k) * N_TOT + n0 + tn];   // coalesced over N
  }
  __syncthreads();
  #pragma unroll
  for (int p = 0; p < 4; ++p) {
    const int n = tk + p * 8;
    const float v = T[tn][n];                            // (k = k0+tn, n = n0+n)
    const bf16_t h = (bf16_t)v;
    const bf16_t l = (bf16_t)(v - (float)h);
    Wth[(size_t)(n0 + n) * K_TOT + k0 + tn] = h;         // coalesced over K
    Wtl[(size_t)(n0 + n) * K_TOT + k0 + tn] = l;
  }
}

// =================================================================
// Main GEMM (async, double-buffered): bf16 hi/lo split WMMA + Yat epilogue
// =================================================================
__global__ __launch_bounds__(256) void yat_gemm_async(const bf16_t* __restrict__ Xh,
                                                      const bf16_t* __restrict__ Xl,
                                                      const bf16_t* __restrict__ Wth,
                                                      const bf16_t* __restrict__ Wtl,
                                                      const float* __restrict__ bias,
                                                      const float* __restrict__ alpha,
                                                      const float* __restrict__ x2,
                                                      const float* __restrict__ w2,
                                                      float* __restrict__ out) {
  __shared__ __align__(16) bf16_t Ah[2][BM][BK];
  __shared__ __align__(16) bf16_t Al[2][BM][BK];
  __shared__ __align__(16) bf16_t Bh[2][BN][BK];
  __shared__ __align__(16) bf16_t Bl[2][BN][BK];

  const int tid   = threadIdx.x;
  const int lane  = tid & 31;
  const int wave  = tid >> 5;
  const int lr    = lane & 15;
  const int lh    = lane >> 4;
  const int waveM = wave & 3;
  const int waveN = wave >> 2;

  const int bm = blockIdx.y * BM;
  const int bn = blockIdx.x * BN;

  // 16B-segment mapping for async tile staging: 512 segs/matrix, 2 per thread
  const int row0 = tid >> 2;                 // 0..63
  const int off0 = (tid & 3) * 8;            // 0,8,16,24 (elements)
  const int row1 = row0 + 64;                // 64..127
  const int off1 = off0;

  v8f acc[2][4];
  #pragma unroll
  for (int m = 0; m < 2; ++m)
    #pragma unroll
    for (int n = 0; n < 4; ++n) {
      v8f z = {0.f,0.f,0.f,0.f,0.f,0.f,0.f,0.f};
      acc[m][n] = z;
    }

  auto issue_tile = [&](int b, int kt) {
    const int k0 = kt * BK;
    const size_t a0 = (size_t)(bm + row0) * K_TOT + k0 + off0;
    const size_t a1 = (size_t)(bm + row1) * K_TOT + k0 + off1;
    const size_t b0 = (size_t)(bn + row0) * K_TOT + k0 + off0;
    const size_t b1 = (size_t)(bn + row1) * K_TOT + k0 + off1;
    async_copy16(Xh  + a0, &Ah[b][row0][off0]);
    async_copy16(Xh  + a1, &Ah[b][row1][off1]);
    async_copy16(Xl  + a0, &Al[b][row0][off0]);
    async_copy16(Xl  + a1, &Al[b][row1][off1]);
    async_copy16(Wth + b0, &Bh[b][row0][off0]);
    async_copy16(Wth + b1, &Bh[b][row1][off1]);
    async_copy16(Wtl + b0, &Bl[b][row0][off0]);
    async_copy16(Wtl + b1, &Bl[b][row1][off1]);
  };

  issue_tile(0, 0);

  for (int kt = 0; kt < NTILES; ++kt) {
    const int cur = kt & 1;
    if (kt + 1 < NTILES) {
      issue_tile(cur ^ 1, kt + 1);
      // 16 issued: wait until only next tile's 8 remain -> current tile landed
      asm volatile("s_wait_asynccnt 0x8" ::: "memory");
    } else {
      asm volatile("s_wait_asynccnt 0x0" ::: "memory");
    }
    __syncthreads();  // all waves' current-tile data visible in LDS

    // ---- fragments: per-lane K = [lh*8, lh*8+8) U [16+lh*8, ...) ----
    v16bf afh[2], afl[2], bfh[4], bfl[4];
    #pragma unroll
    for (int m = 0; m < 2; ++m) {
      const bf16_t* ph = &Ah[cur][waveM * 32 + m * 16 + lr][0];
      const bf16_t* pl = &Al[cur][waveM * 32 + m * 16 + lr][0];
      v8bf h0 = *(const v8bf*)(ph + lh * 8);
      v8bf h1 = *(const v8bf*)(ph + 16 + lh * 8);
      v8bf l0 = *(const v8bf*)(pl + lh * 8);
      v8bf l1 = *(const v8bf*)(pl + 16 + lh * 8);
      afh[m] = __builtin_shufflevector(h0, h1, 0,1,2,3,4,5,6,7,8,9,10,11,12,13,14,15);
      afl[m] = __builtin_shufflevector(l0, l1, 0,1,2,3,4,5,6,7,8,9,10,11,12,13,14,15);
    }
    #pragma unroll
    for (int n = 0; n < 4; ++n) {
      const bf16_t* ph = &Bh[cur][waveN * 64 + n * 16 + lr][0];
      const bf16_t* pl = &Bl[cur][waveN * 64 + n * 16 + lr][0];
      v8bf h0 = *(const v8bf*)(ph + lh * 8);
      v8bf h1 = *(const v8bf*)(ph + 16 + lh * 8);
      v8bf l0 = *(const v8bf*)(pl + lh * 8);
      v8bf l1 = *(const v8bf*)(pl + 16 + lh * 8);
      bfh[n] = __builtin_shufflevector(h0, h1, 0,1,2,3,4,5,6,7,8,9,10,11,12,13,14,15);
      bfl[n] = __builtin_shufflevector(l0, l1, 0,1,2,3,4,5,6,7,8,9,10,11,12,13,14,15);
    }

    #pragma unroll
    for (int m = 0; m < 2; ++m)
      #pragma unroll
      for (int n = 0; n < 4; ++n) {
        acc[m][n] = __builtin_amdgcn_wmma_f32_16x16x32_bf16(
            false, afh[m], false, bfh[n], (short)0, acc[m][n], false, false);
        acc[m][n] = __builtin_amdgcn_wmma_f32_16x16x32_bf16(
            false, afh[m], false, bfl[n], (short)0, acc[m][n], false, false);
        acc[m][n] = __builtin_amdgcn_wmma_f32_16x16x32_bf16(
            false, afl[m], false, bfh[n], (short)0, acc[m][n], false, false);
      }
    __syncthreads();  // done reading buf[cur] before it is re-filled
  }

  // ---- Yat epilogue ----
  const float s0    = sqrtf((float)N_TOT) / logf(1.0f + (float)N_TOT);
  const float scale = powf(s0, alpha[0]);
  #pragma unroll
  for (int m = 0; m < 2; ++m) {
    const int rowb = bm + waveM * 32 + m * 16 + lh * 8;
    float xs[8];
    #pragma unroll
    for (int r = 0; r < 8; ++r) xs[r] = x2[rowb + r];
    #pragma unroll
    for (int n = 0; n < 4; ++n) {
      const int col = bn + waveN * 64 + n * 16 + lr;
      const float wsum = w2[col];
      const float bv   = bias[col];
      #pragma unroll
      for (int r = 0; r < 8; ++r) {
        float y = acc[m][n][r];
        float d = xs[r] + wsum - 2.0f * y + EPSV;
        out[(size_t)(rowb + r) * N_TOT + col] = (y * y / d + bv) * scale;
      }
    }
  }
}

// =================================================================
// Fallback GEMM: in-loop f32->bf16 conversion (small workspace)
// =================================================================
__global__ __launch_bounds__(256) void yat_gemm_fused(const float* __restrict__ A,
                                                      const float* __restrict__ Bm,
                                                      const float* __restrict__ bias,
                                                      const float* __restrict__ alpha,
                                                      const float* __restrict__ x2,
                                                      const float* __restrict__ w2,
                                                      float* __restrict__ out) {
  __shared__ __align__(16) bf16_t Ah[BM][BK];
  __shared__ __align__(16) bf16_t Al[BM][BK];
  __shared__ __align__(16) bf16_t Bh[BN][BK];
  __shared__ __align__(16) bf16_t Bl[BN][BK];

  const int tid   = threadIdx.x;
  const int lane  = tid & 31;
  const int wave  = tid >> 5;
  const int lr    = lane & 15;
  const int lh    = lane >> 4;
  const int waveM = wave & 3;
  const int waveN = wave >> 2;
  const int bm = blockIdx.y * BM;
  const int bn = blockIdx.x * BN;

  v8f acc[2][4];
  #pragma unroll
  for (int m = 0; m < 2; ++m)
    #pragma unroll
    for (int n = 0; n < 4; ++n) {
      v8f z = {0.f,0.f,0.f,0.f,0.f,0.f,0.f,0.f};
      acc[m][n] = z;
    }

  const int arow = tid >> 3;
  const int acol = (tid & 7) * 4;
  const int bcol  = tid & 127;
  const int bkrow = (tid >> 7) * 4;

  for (int k0 = 0; k0 < K_TOT; k0 += BK) {
    #pragma unroll
    for (int p = 0; p < 4; ++p) {
      const int r = arow + p * 32;
      v4f a = *(const v4f*)(A + (size_t)(bm + r) * K_TOT + k0 + acol);
      v4bf h, l;
      #pragma unroll
      for (int i = 0; i < 4; ++i) {
        bf16_t hh = (bf16_t)a[i];
        h[i] = hh;
        l[i] = (bf16_t)(a[i] - (float)hh);
      }
      *(v4bf*)&Ah[r][acol] = h;
      *(v4bf*)&Al[r][acol] = l;
    }
    #pragma unroll
    for (int p = 0; p < 4; ++p) {
      const int kk = p * 8 + bkrow;
      v4bf h, l;
      #pragma unroll
      for (int i = 0; i < 4; ++i) {
        float v = Bm[(size_t)(k0 + kk + i) * N_TOT + bn + bcol];
        bf16_t hh = (bf16_t)v;
        h[i] = hh;
        l[i] = (bf16_t)(v - (float)hh);
      }
      *(v4bf*)&Bh[bcol][kk] = h;
      *(v4bf*)&Bl[bcol][kk] = l;
    }
    __syncthreads();

    v16bf afh[2], afl[2], bfh[4], bfl[4];
    #pragma unroll
    for (int m = 0; m < 2; ++m) {
      const bf16_t* ph = &Ah[waveM * 32 + m * 16 + lr][0];
      const bf16_t* pl = &Al[waveM * 32 + m * 16 + lr][0];
      v8bf h0 = *(const v8bf*)(ph + lh * 8);
      v8bf h1 = *(const v8bf*)(ph + 16 + lh * 8);
      v8bf l0 = *(const v8bf*)(pl + lh * 8);
      v8bf l1 = *(const v8bf*)(pl + 16 + lh * 8);
      afh[m] = __builtin_shufflevector(h0, h1, 0,1,2,3,4,5,6,7,8,9,10,11,12,13,14,15);
      afl[m] = __builtin_shufflevector(l0, l1, 0,1,2,3,4,5,6,7,8,9,10,11,12,13,14,15);
    }
    #pragma unroll
    for (int n = 0; n < 4; ++n) {
      const bf16_t* ph = &Bh[waveN * 64 + n * 16 + lr][0];
      const bf16_t* pl = &Bl[waveN * 64 + n * 16 + lr][0];
      v8bf h0 = *(const v8bf*)(ph + lh * 8);
      v8bf h1 = *(const v8bf*)(ph + 16 + lh * 8);
      v8bf l0 = *(const v8bf*)(pl + lh * 8);
      v8bf l1 = *(const v8bf*)(pl + 16 + lh * 8);
      bfh[n] = __builtin_shufflevector(h0, h1, 0,1,2,3,4,5,6,7,8,9,10,11,12,13,14,15);
      bfl[n] = __builtin_shufflevector(l0, l1, 0,1,2,3,4,5,6,7,8,9,10,11,12,13,14,15);
    }
    #pragma unroll
    for (int m = 0; m < 2; ++m)
      #pragma unroll
      for (int n = 0; n < 4; ++n) {
        acc[m][n] = __builtin_amdgcn_wmma_f32_16x16x32_bf16(
            false, afh[m], false, bfh[n], (short)0, acc[m][n], false, false);
        acc[m][n] = __builtin_amdgcn_wmma_f32_16x16x32_bf16(
            false, afh[m], false, bfl[n], (short)0, acc[m][n], false, false);
        acc[m][n] = __builtin_amdgcn_wmma_f32_16x16x32_bf16(
            false, afl[m], false, bfh[n], (short)0, acc[m][n], false, false);
      }
    __syncthreads();
  }

  const float s0    = sqrtf((float)N_TOT) / logf(1.0f + (float)N_TOT);
  const float scale = powf(s0, alpha[0]);
  #pragma unroll
  for (int m = 0; m < 2; ++m) {
    const int rowb = bm + waveM * 32 + m * 16 + lh * 8;
    float xs[8];
    #pragma unroll
    for (int r = 0; r < 8; ++r) xs[r] = x2[rowb + r];
    #pragma unroll
    for (int n = 0; n < 4; ++n) {
      const int col = bn + waveN * 64 + n * 16 + lr;
      const float wsum = w2[col];
      const float bv   = bias[col];
      #pragma unroll
      for (int r = 0; r < 8; ++r) {
        float y = acc[m][n][r];
        float d = xs[r] + wsum - 2.0f * y + EPSV;
        out[(size_t)(rowb + r) * N_TOT + col] = (y * y / d + bv) * scale;
      }
    }
  }
}

extern "C" void kernel_launch(void* const* d_in, const int* in_sizes, int n_in,
                              void* d_out, int out_size, void* d_ws, size_t ws_size,
                              hipStream_t stream) {
  const float* X     = (const float*)d_in[0];
  const float* W     = (const float*)d_in[1];
  const float* bias  = (const float*)d_in[2];
  const float* alpha = (const float*)d_in[3];
  float* out = (float*)d_out;

  float* x2 = (float*)d_ws;
  float* w2 = x2 + M_TOT;

  const size_t bytes_norms = (size_t)(M_TOT + N_TOT) * sizeof(float);
  const size_t bytes_X = (size_t)M_TOT * K_TOT * 2 * sizeof(bf16_t);  // hi+lo
  const size_t bytes_W = (size_t)N_TOT * K_TOT * 2 * sizeof(bf16_t);  // hi+lo
  const size_t need    = bytes_norms + bytes_X + bytes_W;

  dim3 grid(N_TOT / BN, M_TOT / BM);

  if (ws_size >= need) {
    bf16_t* Xh  = (bf16_t*)((char*)d_ws + bytes_norms);
    bf16_t* Xl  = Xh + (size_t)M_TOT * K_TOT;
    bf16_t* Wth = Xl + (size_t)M_TOT * K_TOT;
    bf16_t* Wtl = Wth + (size_t)N_TOT * K_TOT;

    yat_rownorm_convert<<<M_TOT, 256, 0, stream>>>(X, Xh, Xl, x2);
    yat_col_sq<<<N_TOT / 256, 256, 0, stream>>>(W, w2);
    dim3 tg(N_TOT / 32, K_TOT / 32);
    yat_transpose_convert<<<tg, 256, 0, stream>>>(W, Wth, Wtl);
    yat_gemm_async<<<grid, 256, 0, stream>>>(Xh, Xl, Wth, Wtl, bias, alpha, x2, w2, out);
  } else {
    yat_row_sq<<<M_TOT, 256, 0, stream>>>(X, x2);
    yat_col_sq<<<N_TOT / 256, 256, 0, stream>>>(W, w2);
    yat_gemm_fused<<<grid, 256, 0, stream>>>(X, W, bias, alpha, x2, w2, out);
  }
}
